// BitLinear_16793322127752
// MI455X (gfx1250) — compile-verified, matrix operand order checked by reading, not run on hardware
//
#include <hip/hip_runtime.h>

typedef __attribute__((ext_vector_type(16))) _Float16 v16h;
typedef __attribute__((ext_vector_type(8)))  _Float16 v8h;
typedef __attribute__((ext_vector_type(8)))  float    v8f;
typedef __attribute__((ext_vector_type(4)))  float    v4f;
typedef __attribute__((ext_vector_type(4)))  int      v4i;

#define K_DIM 2048
#define N_DIM 2048
#define TM 128
#define TN 128
#define TK 32
#define LDSS 40            // padded LDS row stride in halves: 80B = 5*16B (keeps b128 aligned)
#define KITERS (K_DIM / TK)

#if defined(__gfx1250__) && __has_builtin(__builtin_amdgcn_global_load_async_to_lds_b128)
#define USE_ASYNC_LDS 1
#else
#define USE_ASYNC_LDS 0
#endif

// ---------------- Stage 1: deterministic partial |W| sums ----------------
__global__ __launch_bounds__(256) void abssum_partial(const float* __restrict__ w,
                                                      float* __restrict__ partial, int n) {
    __shared__ float red[256];
    int tid = threadIdx.x;
    float s = 0.0f;
    for (int i = blockIdx.x * 256 + tid; i < n; i += gridDim.x * 256)
        s += fabsf(w[i]);
    red[tid] = s;
    __syncthreads();
    for (int off = 128; off > 0; off >>= 1) {
        if (tid < off) red[tid] += red[tid + off];
        __syncthreads();
    }
    if (tid == 0) partial[blockIdx.x] = red[0];
}

// ---------------- Stage 2: finalize threshold (single block, fixed order) ----------------
__global__ __launch_bounds__(256) void finalize_thr(const float* __restrict__ partial,
                                                    float* __restrict__ thr_out) {
    __shared__ float red[256];
    int tid = threadIdx.x;
    float s = 0.0f;
    for (int i = tid; i < 1024; i += 256) s += partial[i];
    red[tid] = s;
    __syncthreads();
    for (int off = 128; off > 0; off >>= 1) {
        if (tid < off) red[tid] += red[tid + off];
        __syncthreads();
    }
    if (tid == 0) {
        float mean  = red[0] / (float)(N_DIM * K_DIM);
        float scale = fmaxf(mean, 1e-6f);
        thr_out[0]  = 0.05f * scale;
    }
}

// ---------------- Stage 3: ternarize W -> fp16 {-1,0,+1} ----------------
__global__ __launch_bounds__(256) void ternarize(const float* __restrict__ w,
                                                 const float* __restrict__ thr_p,
                                                 _Float16* __restrict__ q, int n) {
    float thr = thr_p[0];
    int i = blockIdx.x * 256 + threadIdx.x;
    if (i < n) {
        float v = w[i];
        float t = (v > thr) ? 1.0f : ((v < -thr) ? -1.0f : 0.0f);
        q[i] = (_Float16)t;
    }
}

// ---------------- Stage 4: WMMA GEMM  Y = X * Wq^T + b ----------------
// Double-buffered LDS, one barrier per K-iteration. B tile travels on the
// CDNA5 async global->LDS path (ASYNCcnt); A tile is f32->f16 converted
// through registers (async moves raw bytes, so it cannot convert).
__global__ __launch_bounds__(256) void bitlinear_wmma(const float* __restrict__ x,
                                                      const _Float16* __restrict__ qw,
                                                      const float* __restrict__ bias,
                                                      float* __restrict__ out, int M) {
    __shared__ __align__(16) _Float16 sA[2][TM][LDSS];
    __shared__ __align__(16) _Float16 sB[2][TN][LDSS];

    const int tid  = threadIdx.x;
    const int lane = tid & 31;
    const int wave = tid >> 5;
    const int wm   = wave & 1;    // wave M index (0..1), 64 rows each
    const int wn   = wave >> 1;   // wave N index (0..3), 32 cols each
    const int r16  = lane & 15;
    const int half = lane >> 4;

    const int blockM = blockIdx.y * TM;
    const int blockN = blockIdx.x * TN;

    // A tile mapping: 128 rows x 32 f32; 8 quads/row, 32 rows/pass, 4 passes.
    const int arow  = tid >> 3;   // 0..31
    const int aquad = tid & 7;    // 0..7

    v8f acc[4][2];
#pragma unroll
    for (int mi = 0; mi < 4; ++mi)
#pragma unroll
        for (int ni = 0; ni < 2; ++ni)
            acc[mi][ni] = (v8f){};

    v4f areg[4];

    auto loadA = [&](int k0) {
#pragma unroll
        for (int p = 0; p < 4; ++p) {
            int row = arow + p * 32;
            areg[p] = *(const v4f*)(x + (size_t)(blockM + row) * K_DIM + k0 + aquad * 4);
        }
    };
    auto storeA = [&](int buf) {
#pragma unroll
        for (int p = 0; p < 4; ++p) {
            int row = arow + p * 32;
            _Float16* dst = &sA[buf][row][aquad * 4];
            dst[0] = (_Float16)areg[p].x;
            dst[1] = (_Float16)areg[p].y;
            dst[2] = (_Float16)areg[p].z;
            dst[3] = (_Float16)areg[p].w;
        }
    };

#if USE_ASYNC_LDS
    // B tile: 128 rows x 32 halves = 512 b128 chunks; 2 chunks per thread,
    // moved global->LDS asynchronously (no VGPR staging).
    auto loadB_async = [&](int k0, int buf) {
#pragma unroll
        for (int t = 0; t < 2; ++t) {
            int cf  = tid + t * 256;
            int row = cf >> 2;    // 4 chunks per row
            int cq  = cf & 3;
            const _Float16* src = qw + (size_t)(blockN + row) * K_DIM + k0 + cq * 8;
            _Float16* dst = &sB[buf][row][cq * 8];
            __builtin_amdgcn_global_load_async_to_lds_b128(
                (__attribute__((address_space(1))) v4i*)src,
                (__attribute__((address_space(3))) v4i*)dst, 0, 0);
        }
    };
    auto asyncWait = [&]() {
#if __has_builtin(__builtin_amdgcn_s_wait_asynccnt)
        __builtin_amdgcn_s_wait_asynccnt(0);
#else
        asm volatile("s_wait_asynccnt 0x0" ::: "memory");
#endif
    };
#else
    v8h breg[2];
    auto loadB = [&](int k0) {
#pragma unroll
        for (int t = 0; t < 2; ++t) {
            int cf  = tid + t * 256;
            int row = cf >> 2;
            int cq  = cf & 3;
            breg[t] = *(const v8h*)(qw + (size_t)(blockN + row) * K_DIM + k0 + cq * 8);
        }
    };
    auto storeB = [&](int buf) {
#pragma unroll
        for (int t = 0; t < 2; ++t) {
            int cf  = tid + t * 256;
            int row = cf >> 2;
            int cq  = cf & 3;
            *(v8h*)(&sB[buf][row][cq * 8]) = breg[t];
        }
    };
    auto asyncWait = [&]() {};
#endif

    auto compute = [&](int buf) {
        // ISA 16-bit A/B layout: lane = half*16 + idx16; two 8-half chunks at
        // K = half*8 and K = 16 + half*8 (TK == 32: exactly one WMMA K-step).
        union Frag { v16h v; v8h h[2]; } a[4], b[2];
#pragma unroll
        for (int mi = 0; mi < 4; ++mi) {
            int row = wm * 64 + mi * 16 + r16;
            a[mi].h[0] = *(const v8h*)(&sA[buf][row][half * 8]);
            a[mi].h[1] = *(const v8h*)(&sA[buf][row][16 + half * 8]);
        }
#pragma unroll
        for (int ni = 0; ni < 2; ++ni) {
            int col = wn * 32 + ni * 16 + r16;
            b[ni].h[0] = *(const v8h*)(&sB[buf][col][half * 8]);
            b[ni].h[1] = *(const v8h*)(&sB[buf][col][16 + half * 8]);
        }
#pragma unroll
        for (int mi = 0; mi < 4; ++mi)
#pragma unroll
            for (int ni = 0; ni < 2; ++ni)
                acc[mi][ni] = __builtin_amdgcn_wmma_f32_16x16x32_f16(
                    false, a[mi].v, false, b[ni].v,
                    (short)0, acc[mi][ni], false, false);
    };

    // ---- Prologue: tile 0 -> buffer 0 ----
    loadA(0);
#if USE_ASYNC_LDS
    loadB_async(0, 0);
#else
    loadB(0);
    storeB(0);
#endif
    storeA(0);
    asyncWait();
    __syncthreads();

    // ---- Pipelined main loop: prefetch k+1 while computing k ----
    for (int it = 0; it < KITERS; ++it) {
        const int p = it & 1;
        const int q = p ^ 1;
        const bool hasNext = (it + 1) < KITERS;
        const int kn = (it + 1) * TK;

        if (hasNext) {
            loadA(kn);
#if USE_ASYNC_LDS
            loadB_async(kn, q);
#else
            loadB(kn);
#endif
        }

        compute(p);

        if (hasNext) {
            storeA(q);
#if !USE_ASYNC_LDS
            storeB(q);
#endif
        }
        asyncWait();
        __syncthreads();
    }

    // ---- Epilogue: C layout is VGPR r -> (M = 8*half + r, N = lane%16) ----
#pragma unroll
    for (int ni = 0; ni < 2; ++ni) {
        int n = blockN + wn * 32 + ni * 16 + r16;
        float bv = bias[n];
#pragma unroll
        for (int mi = 0; mi < 4; ++mi) {
            int m0 = blockM + wm * 64 + mi * 16 + half * 8;
#pragma unroll
            for (int r = 0; r < 8; ++r) {
                out[(size_t)(m0 + r) * N_DIM + n] = acc[mi][ni][r] + bv;
            }
        }
    }
}

extern "C" void kernel_launch(void* const* d_in, const int* in_sizes, int n_in,
                              void* d_out, int out_size, void* d_ws, size_t ws_size,
                              hipStream_t stream) {
    const float* x    = (const float*)d_in[0];   // [4,4096,2048] f32
    const float* w    = (const float*)d_in[1];   // [2048,2048]   f32
    const float* bias = (const float*)d_in[2];   // [2048]        f32
    float*       out  = (float*)d_out;           // [4,4096,2048] f32

    const int M  = in_sizes[0] / K_DIM;          // 16384
    const int NK = N_DIM * K_DIM;                // 4,194,304

    // Workspace layout (all written before read; no dependence on poison):
    //   [0, 4096)      : 1024 block partial sums (f32)
    //   [4096, 4100)   : threshold scalar (f32)
    //   [8192, +8MB)   : ternarized weights (f16, [N][K] row-major, 16B aligned)
    float*    partial = (float*)d_ws;
    float*    thr     = (float*)((char*)d_ws + 4096);
    _Float16* qw      = (_Float16*)((char*)d_ws + 8192);

    abssum_partial<<<1024, 256, 0, stream>>>(w, partial, NK);
    finalize_thr<<<1, 256, 0, stream>>>(partial, thr);
    ternarize<<<NK / 256, 256, 0, stream>>>(w, thr, qw, NK);

    dim3 grid(N_DIM / TN, M / TM);               // 16 x 128 = 2048 blocks
    bitlinear_wmma<<<grid, 256, 0, stream>>>(x, qw, bias, out, M);
}
